// TokenMixer_50345606644161
// MI455X (gfx1250) — matrix operand order repaired and checked by reading, not compiled, not run on hardware
//
#include <hip/hip_runtime.h>
#include <hip/hip_bf16.h>

typedef __attribute__((ext_vector_type(16))) __bf16 v16bf;
typedef __attribute__((ext_vector_type(8)))  __bf16 v8bf;
typedef __attribute__((ext_vector_type(4)))  __bf16 v4bf;
typedef __attribute__((ext_vector_type(8)))  float  v8f;

#define B_    4
#define S_    1024
#define D_    1024
#define H_    4096
#define KTAP  5

// workspace layout (bytes)
#define OFF_XN  (0ull)                         // bf16 [B][S][D]          8 MB
#define OFF_GT  (8388608ull)                   // bf16 [B][S][H]         32 MB
#define OFF_W1  (41943040ull)                  // bf16 [K][H][D]         40 MB
#define OFF_W2  (83886080ull)                  // bf16 [K][S][H]         40 MB

// ---------------------------------------------------------------------------
// Async global -> LDS staging (CDNA5 GLOBAL_LOAD_ASYNC_TO_LDS_B128, ASYNCcnt),
// with a synchronous fallback so the file always compiles.
// Builtin signature (from hipcc diagnostic): param0 is
//   'int __vector(4) __device__ *'  == addrspace(1) int4-vector pointer.
// ---------------------------------------------------------------------------
#if defined(__has_builtin)
#if __has_builtin(__builtin_amdgcn_global_load_async_to_lds_b128)
#define HAVE_ASYNC_LDS 1
#endif
#endif

typedef int v4i_vec __attribute__((vector_size(4 * sizeof(int))));
typedef __attribute__((address_space(1))) v4i_vec* gptr_b128;
typedef __attribute__((address_space(3))) v4i_vec* lptr_b128;

__device__ __forceinline__ void g2l_b128(const void* g, void* l) {
#if defined(HAVE_ASYNC_LDS)
  __builtin_amdgcn_global_load_async_to_lds_b128((gptr_b128)g, (lptr_b128)l, 0, 0);
#else
  *(uint4*)l = *(const uint4*)g;
#endif
}

__device__ __forceinline__ void g2l_b64(const void* g, void* l, int bytes /*64*/) {
  // copy 64 bytes from global to LDS as 4 x b128
  #pragma unroll
  for (int o = 0; o < 64; o += 16)
    g2l_b128((const char*)g + o, (char*)l + o);
  (void)bytes;
}

__device__ __forceinline__ void wait_async_lds() {
#if defined(__has_builtin)
#if __has_builtin(__builtin_amdgcn_s_wait_asynccnt)
  __builtin_amdgcn_s_wait_asynccnt(0);
  return;
#elif defined(HAVE_ASYNC_LDS)
  asm volatile("s_wait_asynccnt 0x0" ::: "memory");
  return;
#endif
#endif
}

// ---------------------------------------------------------------------------
// w[co][ci][k] fp32  ->  o[k][co][ci] bf16   (per-tap GEMM A operand)
// ---------------------------------------------------------------------------
__global__ void reorg_weights_kernel(const float* __restrict__ w,
                                     __bf16* __restrict__ o,
                                     int cout, int cin) {
  const int idx = blockIdx.x * blockDim.x + threadIdx.x;   // over cout*cin
  const int total = cout * cin;
  if (idx >= total) return;
  const float* src = w + (size_t)idx * KTAP;               // k contiguous
  #pragma unroll
  for (int k = 0; k < KTAP; ++k)
    o[(size_t)k * total + idx] = (__bf16)src[k];
}

// ---------------------------------------------------------------------------
// LayerNorm over D, bf16 output in natural [b][s][d] layout (= GEMM-B layout)
// ---------------------------------------------------------------------------
__global__ __launch_bounds__(256)
void layernorm_bf16_kernel(const float* __restrict__ x,
                           const float* __restrict__ gamma,
                           const float* __restrict__ beta,
                           __bf16* __restrict__ xn) {
  const int row = blockIdx.x;                 // b*S + s
  const int t   = threadIdx.x;                // 256 threads, 4 floats each
  const float* xr = x + (size_t)row * D_;

  float4 v = ((const float4*)xr)[t];
  float s  = v.x + v.y + v.z + v.w;
  float sq = v.x * v.x + v.y * v.y + v.z * v.z + v.w * v.w;

  // wave32 reduction
  #pragma unroll
  for (int o = 16; o > 0; o >>= 1) {
    s  += __shfl_down(s,  o, 32);
    sq += __shfl_down(sq, o, 32);
  }
  __shared__ float ps[8], pq[8];
  const int wid = t >> 5, lane = t & 31;
  if (lane == 0) { ps[wid] = s; pq[wid] = sq; }
  __syncthreads();
  if (t == 0) {
    float a = 0.f, b = 0.f;
    #pragma unroll
    for (int i = 0; i < 8; ++i) { a += ps[i]; b += pq[i]; }
    ps[0] = a; pq[0] = b;
  }
  __syncthreads();
  const float mu  = ps[0] * (1.0f / D_);
  const float var = pq[0] * (1.0f / D_) - mu * mu;
  const float rs  = rsqrtf(var + 1e-5f);

  const float4 g  = ((const float4*)gamma)[t];
  const float4 bb = ((const float4*)beta)[t];
  v4bf o;
  o[0] = (__bf16)((v.x - mu) * rs * g.x + bb.x);
  o[1] = (__bf16)((v.y - mu) * rs * g.y + bb.y);
  o[2] = (__bf16)((v.z - mu) * rs * g.z + bb.z);
  o[3] = (__bf16)((v.w - mu) * rs * g.w + bb.w);
  *(v4bf*)(xn + (size_t)row * D_ + t * 4) = o;
}

// ---------------------------------------------------------------------------
// Shifted-GEMM conv1d:  out[m, l] = sum_tap sum_c A[tap][m][c] * Bact[l+tap-2][c]
// Tile 128x128, K-step 64, double-buffered LDS fed by async global->LDS copies.
// 8 waves (2 x 4); wave computes 64x32 = 8 accumulators; 16 WMMAs per barrier.
// EPI==0: +bias, exact GELU, bf16 store to g_t[b][l][Mtot]
// EPI==1: +bias, +x residual, fp32 store to out[b][l][Mtot]
// ---------------------------------------------------------------------------
template <int EPI, int CIN>
__global__ __launch_bounds__(256)
void conv_gemm_kernel(const __bf16* __restrict__ A,     // [KTAP][Mtot][CIN]
                      const __bf16* __restrict__ Bact,  // [B][S][CIN]
                      const float*  __restrict__ bias,  // [Mtot]
                      void* __restrict__ outp,
                      const float* __restrict__ xres,   // EPI==1 only
                      int Mtot) {
  constexpr int NC  = CIN / 64;        // k-iterations per tap
  constexpr int NIT = KTAP * NC;       // total pipeline iterations (even)

  const int bz = blockIdx.z;
  const int m0 = blockIdx.y * 128;
  const int l0 = blockIdx.x * 128;

  __shared__ __align__(128) __bf16 At[2][128][64];   // 2 x 16 KB  [m][k]
  __shared__ __align__(128) __bf16 Bt[2][128][64];   // 2 x 16 KB  [n][k]
  __shared__ float biasS[128];

  const int tid  = threadIdx.x;
  const int lane = tid & 31;
  const int wid  = tid >> 5;
  const int wm   = wid & 1;            // 2 waves over M (64 rows each)
  const int wn   = wid >> 1;           // 4 waves over N (32 cols each)

  if (tid < 128) biasS[tid] = bias[m0 + tid];

  v8f acc[4][2];
  const v8f vzero = {0.f, 0.f, 0.f, 0.f, 0.f, 0.f, 0.f, 0.f};
  #pragma unroll
  for (int fm = 0; fm < 4; ++fm)
    #pragma unroll
    for (int fn = 0; fn < 2; ++fn) acc[fm][fn] = vzero;

  const int rowT = tid >> 1;           // 0..127 : tile row staged by this thread
  const int colT = (tid & 1) * 32;     // 0/32   : 64-byte half-row

  // stage tiles for pipeline iteration `it` into buffer `buf`
  auto issue = [&](int it, int buf) {
    const int tap = it / NC;           // NC is a power of two -> shift
    const int c0  = (it - tap * NC) * 64;
    const __bf16* gA =
        A + (size_t)tap * Mtot * CIN + (size_t)(m0 + rowT) * CIN + c0 + colT;
    g2l_b64(gA, &At[buf][rowT][colT], 64);
    const int l_in = l0 + rowT + tap - 2;          // shifted, zero-padded
    if (l_in >= 0 && l_in < S_) {
      const __bf16* gB = Bact + ((size_t)bz * S_ + l_in) * CIN + c0 + colT;
      g2l_b64(gB, &Bt[buf][rowT][colT], 64);
    } else {
      const uint4 z4 = {0u, 0u, 0u, 0u};
      #pragma unroll
      for (int o = 0; o < 32; o += 8) *(uint4*)(&Bt[buf][rowT][colT + o]) = z4;
    }
  };

  issue(0, 0);                         // prologue: prefetch first tile

  #pragma unroll 2
  for (int it = 0; it < NIT; ++it) {
    const int buf = it & 1;
    wait_async_lds();                  // this wave's copies into `buf` done
    __syncthreads();                   // whole tile visible; prev compute done
    if (it + 1 < NIT) issue(it + 1, buf ^ 1);   // overlap copy with compute

    #pragma unroll
    for (int ks = 0; ks < 2; ++ks) {
      const int kb  = ks * 32;
      // A: lanes 0-15 hold K{0-7,16-23}; lanes 16-31 hold K{8-15,24-31}
      const int ksA = kb + (lane >> 4) * 8;
      // B: lanes 0-15 hold K{0-7,16-23}; lanes 16-31 hold K{8-15,24-31}
      const int ksB = kb + (lane >> 4) * 8;
      v16bf afrag[4];
      #pragma unroll
      for (int fm = 0; fm < 4; ++fm) {
        const int m = wm * 64 + fm * 16 + (lane & 15);
        const v8bf lo = *(const v8bf*)(&At[buf][m][ksA]);
        const v8bf hi = *(const v8bf*)(&At[buf][m][ksA + 16]);
        #pragma unroll
        for (int i = 0; i < 8; ++i) { afrag[fm][i] = lo[i]; afrag[fm][i + 8] = hi[i]; }
      }
      v16bf bfrag[2];
      #pragma unroll
      for (int fn = 0; fn < 2; ++fn) {
        const int n = wn * 32 + fn * 16 + (lane & 15);
        const v8bf blo = *(const v8bf*)(&Bt[buf][n][ksB]);
        const v8bf bhi = *(const v8bf*)(&Bt[buf][n][ksB + 16]);
        #pragma unroll
        for (int i = 0; i < 8; ++i) { bfrag[fn][i] = blo[i]; bfrag[fn][i + 8] = bhi[i]; }
      }
      #pragma unroll
      for (int fm = 0; fm < 4; ++fm)
        #pragma unroll
        for (int fn = 0; fn < 2; ++fn)
          acc[fm][fn] = __builtin_amdgcn_wmma_f32_16x16x32_bf16(
              false, afrag[fm], false, bfrag[fn],
              (short)0, acc[fm][fn], false, false);
    }
  }

  // ---- epilogue: D layout => lane owns col n, 8 consecutive rows m ----
  const int nlane = lane & 15;
  const int mhalf = (lane >> 4) * 8;
  #pragma unroll
  for (int fm = 0; fm < 4; ++fm) {
    const int mb = wm * 64 + fm * 16 + mhalf;          // 8 contiguous m rows
    #pragma unroll
    for (int fn = 0; fn < 2; ++fn) {
      const int n = wn * 32 + fn * 16 + nlane;
      const size_t rowoff =
          ((size_t)bz * S_ + (size_t)(l0 + n)) * (size_t)Mtot + (size_t)(m0 + mb);
      if (EPI == 0) {
        v8bf o;
        #pragma unroll
        for (int i = 0; i < 8; ++i) {
          float v = acc[fm][fn][i] + biasS[mb + i];
          v = 0.5f * v * (1.0f + erff(v * 0.70710678118654752440f));  // exact GELU
          o[i] = (__bf16)v;
        }
        *(v8bf*)((__bf16*)outp + rowoff) = o;          // one b128, contiguous in h
      } else {
        const float* xr = xres + rowoff;
        const float4 x0 = *(const float4*)(xr);
        const float4 x1 = *(const float4*)(xr + 4);
        float4 r0, r1;
        r0.x = acc[fm][fn][0] + biasS[mb + 0] + x0.x;
        r0.y = acc[fm][fn][1] + biasS[mb + 1] + x0.y;
        r0.z = acc[fm][fn][2] + biasS[mb + 2] + x0.z;
        r0.w = acc[fm][fn][3] + biasS[mb + 3] + x0.w;
        r1.x = acc[fm][fn][4] + biasS[mb + 4] + x1.x;
        r1.y = acc[fm][fn][5] + biasS[mb + 5] + x1.y;
        r1.z = acc[fm][fn][6] + biasS[mb + 6] + x1.z;
        r1.w = acc[fm][fn][7] + biasS[mb + 7] + x1.w;
        *(float4*)((float*)outp + rowoff)     = r0;
        *(float4*)((float*)outp + rowoff + 4) = r1;
      }
    }
  }
}

// ---------------------------------------------------------------------------
extern "C" void kernel_launch(void* const* d_in, const int* in_sizes, int n_in,
                              void* d_out, int out_size, void* d_ws, size_t ws_size,
                              hipStream_t stream) {
  const float* x     = (const float*)d_in[0];   // [B,S,D]
  const float* gamma = (const float*)d_in[1];   // [D]
  const float* beta  = (const float*)d_in[2];   // [D]
  const float* w1    = (const float*)d_in[3];   // [H,D,K]
  const float* b1    = (const float*)d_in[4];   // [H]
  const float* w2    = (const float*)d_in[5];   // [S,H,K]
  const float* b2    = (const float*)d_in[6];   // [S]

  char* ws = (char*)d_ws;
  __bf16* xn  = (__bf16*)(ws + OFF_XN);
  __bf16* gt  = (__bf16*)(ws + OFF_GT);
  __bf16* W1k = (__bf16*)(ws + OFF_W1);
  __bf16* W2k = (__bf16*)(ws + OFF_W2);

  // 1) weights -> per-tap bf16 GEMM A operands
  reorg_weights_kernel<<<(H_ * D_ + 255) / 256, 256, 0, stream>>>(w1, W1k, H_, D_);
  reorg_weights_kernel<<<(S_ * H_ + 255) / 256, 256, 0, stream>>>(w2, W2k, S_, H_);

  // 2) LayerNorm -> bf16 activations in GEMM-B layout
  layernorm_bf16_kernel<<<B_ * S_, 256, 0, stream>>>(x, gamma, beta, xn);

  // 3) conv1 (1024 -> 4096) + bias + exact GELU, bf16 intermediate [b][l][H]
  conv_gemm_kernel<0, D_><<<dim3(S_ / 128, H_ / 128, B_), 256, 0, stream>>>(
      W1k, xn, b1, (void*)gt, nullptr, H_);

  // 4) conv2 (4096 -> 1024) + bias + residual, fp32 output [b][l][D]
  conv_gemm_kernel<1, H_><<<dim3(S_ / 128, S_ / 128, B_), 256, 0, stream>>>(
      W2k, gt, b2, d_out, x, S_);
}
